// CompGCNLayer_5832565588650
// MI455X (gfx1250) — compile-verified, matrix operand order checked by reading, not run on hardware
//
#include <hip/hip_runtime.h>
#include <hip/hip_bf16.h>
#include <stdint.h>

#define DIM 128

typedef __attribute__((ext_vector_type(16))) __bf16 v16bf;
typedef __attribute__((ext_vector_type(8)))  __bf16 v8bf;
typedef __attribute__((ext_vector_type(8)))  float  v8f;

// ---------------------------------------------------------------- zero ws
__global__ void cgcn_zero4(float4* p, long long n4) {
    long long i = (long long)blockIdx.x * blockDim.x + threadIdx.x;
    long long stride = (long long)gridDim.x * blockDim.x;
    float4 z = {0.f, 0.f, 0.f, 0.f};
    for (; i < n4; i += stride) p[i] = z;
}

// ------------------------------------------- concat weights (bf16, /3) + bias
__global__ void cgcn_prep_weights(const float* __restrict__ WO,
                                  const float* __restrict__ WI,
                                  const float* __restrict__ WS,
                                  const float* __restrict__ bO,
                                  const float* __restrict__ bI,
                                  const float* __restrict__ bS,
                                  __bf16* __restrict__ Wcat,
                                  float* __restrict__ bias) {
    int gid = blockIdx.x * blockDim.x + threadIdx.x;
    if (gid < DIM * 3 * DIM) {
        int n = gid / (3 * DIM);
        int k = gid % (3 * DIM);
        const float* W = (k < DIM) ? WO : ((k < 2 * DIM) ? WI : WS);
        float v = W[n * DIM + (k % DIM)] * (1.0f / 3.0f);
        Wcat[(size_t)n * (3 * DIM) + k] = (__bf16)v;   // B[k][n] = W[n][k]/3
    }
    if (gid < DIM) bias[gid] = (bO[gid] + bI[gid] + bS[gid]) * (1.0f / 3.0f);
}

// ---------------------------------------------------------- edge scatter
// 32 lanes per edge, 4 floats per lane (float4 gathers, f32 atomics to L2)
__global__ void cgcn_edge_scatter(const float* __restrict__ node,
                                  const float* __restrict__ edge,
                                  const int* __restrict__ src,
                                  const int* __restrict__ dst,
                                  float* sum_out, float* sum_in,
                                  float* cnt_out, float* cnt_in, int n_edges) {
    long long gid = (long long)blockIdx.x * blockDim.x + threadIdx.x;
    int e    = (int)(gid >> 5);
    int lane = (int)(gid & 31);
    if (e >= n_edges) return;
    int s = src[e], t = dst[e];
    float4 a = ((const float4*)(node + (size_t)s * DIM))[lane];
    float4 b = ((const float4*)(node + (size_t)t * DIM))[lane];
    float4 c = ((const float4*)(edge + (size_t)e * DIM))[lane];
    float* so = sum_out + (size_t)t * DIM + lane * 4;
    float* si = sum_in  + (size_t)s * DIM + lane * 4;
    atomicAdd(so + 0, a.x - c.x); atomicAdd(so + 1, a.y - c.y);
    atomicAdd(so + 2, a.z - c.z); atomicAdd(so + 3, a.w - c.w);
    atomicAdd(si + 0, b.x - c.x); atomicAdd(si + 1, b.y - c.y);
    atomicAdd(si + 2, b.z - c.z); atomicAdd(si + 3, b.w - c.w);
    if (lane == 0) { atomicAdd(cnt_out + t, 1.0f); atomicAdd(cnt_in + s, 1.0f); }
}

// -------------------------------------------------- WMMA fragment loader
// 16-bit operand 16x32: lane holds 8 contiguous bf16 at +0 and 8 at +16
__device__ __forceinline__ v16bf cgcn_load_frag16(const __bf16* p) {
    v8bf lo = *(const v8bf*)(p);
    v8bf hi = *(const v8bf*)(p + 16);
    v16bf r;
#pragma unroll
    for (int i = 0; i < 8; ++i) { r[i] = lo[i]; r[i + 8] = hi[i]; }
    return r;
}

// ------------------------------- fused GEMM (bf16 WMMA, K=384) + BN stats
// 1 block = 16 node rows; 8 waves, each wave owns one 16-col tile of N=128
__global__ __launch_bounds__(256, 2)
void cgcn_gemm_wmma(const float* __restrict__ sum_out,
                    const float* __restrict__ sum_in,
                    const float* __restrict__ cnt_out,
                    const float* __restrict__ cnt_in,
                    const float* __restrict__ node,
                    const __bf16* __restrict__ Wcat,
                    const float* __restrict__ bias,
                    float* __restrict__ out,
                    float* stat_sum, float* stat_sq, int n_nodes) {
    __shared__ __bf16 lds_a[16 * 3 * DIM];   // 16 rows x 384 k, 12 KB

    int tid  = threadIdx.x;
    int base = blockIdx.x * 16;

    // ---- stage A tile (scaled means + self) as bf16 into LDS
    int r  = tid >> 4;            // 0..15 row in tile
    int c8 = (tid & 15) * 8;      // 8-wide k chunk
    int row = base + r;
    int rc  = row < n_nodes ? row : (n_nodes - 1);
    float inv_o = 1.0f / fmaxf(cnt_out[rc], 1.0f);
    float inv_i = 1.0f / fmaxf(cnt_in[rc], 1.0f);
    const float* bases[3] = { sum_out + (size_t)rc * DIM,
                              sum_in  + (size_t)rc * DIM,
                              node    + (size_t)rc * DIM };
    float scl[3] = { inv_o, inv_i, 1.0f };
#pragma unroll
    for (int s = 0; s < 3; ++s) {
        const float4* p = (const float4*)(bases[s] + c8);
        float4 x = p[0], y = p[1];
        v8bf v;
        v[0] = (__bf16)(x.x * scl[s]); v[1] = (__bf16)(x.y * scl[s]);
        v[2] = (__bf16)(x.z * scl[s]); v[3] = (__bf16)(x.w * scl[s]);
        v[4] = (__bf16)(y.x * scl[s]); v[5] = (__bf16)(y.y * scl[s]);
        v[6] = (__bf16)(y.z * scl[s]); v[7] = (__bf16)(y.w * scl[s]);
        *(v8bf*)(&lds_a[r * (3 * DIM) + s * DIM + c8]) = v;
    }
    __syncthreads();

    // ---- 12 x v_wmma_f32_16x16x32_bf16 along K=384
    int wave = tid >> 5;                 // n-tile (0..7)
    int lane = tid & 31;
    int col  = wave * 16 + (lane & 15);  // output column
    int hk   = (lane >> 4) * 8;          // K sub-offset per lane half
    const __bf16* arow = &lds_a[(lane & 15) * (3 * DIM)];
    const __bf16* brow = Wcat + (size_t)col * (3 * DIM);
    v8f acc = {};
#pragma unroll
    for (int ks = 0; ks < 12; ++ks) {
        int k0 = ks * 32 + hk;
        v16bf av = cgcn_load_frag16(arow + k0);
        v16bf bv = cgcn_load_frag16(brow + k0);
        acc = __builtin_amdgcn_wmma_f32_16x16x32_bf16(
            false, av, false, bv, (short)0, acc, false, false);
    }

    // ---- epilogue: bias, store pre-BN h, accumulate BN stats
    float bb = bias[col];
    int mbase = (lane < 16) ? 0 : 8;     // C/D layout: vgpr r -> M=r or r+8
    float s1 = 0.f, s2 = 0.f;
#pragma unroll
    for (int i = 0; i < 8; ++i) {
        int m = base + mbase + i;
        float h = acc[i] + bb;
        if (m < n_nodes) {
            out[(size_t)m * DIM + col] = h;
            s1 += h; s2 += h * h;
        }
    }
    s1 += __shfl_xor(s1, 16, 32);        // combine the two lane halves (wave32)
    s2 += __shfl_xor(s2, 16, 32);
    if (lane < 16) {
        atomicAdd(stat_sum + col, s1);
        atomicAdd(stat_sq  + col, s2);
    }
}

// ------------------------------------------------------- BN finalize
__global__ void cgcn_bn_finalize(const float* __restrict__ stat_sum,
                                 const float* __restrict__ stat_sq,
                                 const float* __restrict__ gamma,
                                 const float* __restrict__ beta,
                                 float* scale, float* shift, float invN) {
    int c = threadIdx.x;
    if (c < DIM) {
        float mean = stat_sum[c] * invN;
        float var  = stat_sq[c] * invN - mean * mean;
        float rs   = rsqrtf(fmaxf(var, 0.f) + 1e-5f);
        float sc   = gamma[c] * rs;
        scale[c] = sc;
        shift[c] = beta[c] - mean * sc;
    }
}

// ------------------------------------------------------- BN apply (in place)
__global__ void cgcn_bn_apply(float* out, const float* __restrict__ scale,
                              const float* __restrict__ shift, long long n4) {
    long long gid = (long long)blockIdx.x * blockDim.x + threadIdx.x;
    if (gid >= n4) return;
    float4 h = ((float4*)out)[gid];
    int c = (int)((gid & 31) << 2);
    h.x = h.x * scale[c + 0] + shift[c + 0];
    h.y = h.y * scale[c + 1] + shift[c + 1];
    h.z = h.z * scale[c + 2] + shift[c + 2];
    h.w = h.w * scale[c + 3] + shift[c + 3];
    ((float4*)out)[gid] = h;
}

static inline size_t align256(size_t x) { return (x + 255) & ~(size_t)255; }

extern "C" void kernel_launch(void* const* d_in, const int* in_sizes, int n_in,
                              void* d_out, int out_size, void* d_ws, size_t ws_size,
                              hipStream_t stream) {
    const float* node_embs = (const float*)d_in[0];
    const float* edge_embs = (const float*)d_in[1];
    const int*   src       = (const int*)d_in[2];
    const int*   dst       = (const int*)d_in[3];
    const float* W_O       = (const float*)d_in[4];
    const float* b_O       = (const float*)d_in[5];
    const float* W_I       = (const float*)d_in[6];
    const float* b_I       = (const float*)d_in[7];
    const float* W_S       = (const float*)d_in[8];
    const float* b_S       = (const float*)d_in[9];
    const float* bn_gamma  = (const float*)d_in[10];
    const float* bn_beta   = (const float*)d_in[11];
    float* out = (float*)d_out;

    const size_t NN = (size_t)in_sizes[0] / DIM;   // 100000
    const size_t NE = (size_t)in_sizes[2];         // 600000

    // workspace layout
    char* w = (char*)d_ws;
    size_t o_sum_out = 0;
    size_t o_sum_in  = o_sum_out + NN * DIM * 4;
    size_t o_cnt_out = o_sum_in  + NN * DIM * 4;
    size_t o_cnt_in  = align256(o_cnt_out + NN * 4);
    size_t o_stats   = align256(o_cnt_in  + NN * 4);          // sum[128], sq[128]
    size_t o_zero_end= align256(o_stats + 1024);
    size_t o_wcat    = o_zero_end;                             // 128*384 bf16
    size_t o_bias    = o_wcat + (size_t)DIM * 3 * DIM * 2;
    size_t o_scale   = o_bias  + 512;
    size_t o_shift   = o_scale + 512;
    (void)o_shift; (void)ws_size;

    float*  sum_out  = (float*)(w + o_sum_out);
    float*  sum_in   = (float*)(w + o_sum_in);
    float*  cnt_out  = (float*)(w + o_cnt_out);
    float*  cnt_in   = (float*)(w + o_cnt_in);
    float*  stat_sum = (float*)(w + o_stats);
    float*  stat_sq  = (float*)(w + o_stats + 512);
    __bf16* Wcat     = (__bf16*)(w + o_wcat);
    float*  bias     = (float*)(w + o_bias);
    float*  scale    = (float*)(w + o_scale);
    float*  shift    = (float*)(w + o_shift);

    // 1) zero accumulators + stats (every call: ws is not re-poisoned)
    long long zero4 = (long long)(o_zero_end / 16);
    cgcn_zero4<<<4096, 256, 0, stream>>>((float4*)w, zero4);

    // 2) weight concat -> bf16 (fold 1/3), combined bias
    cgcn_prep_weights<<<(DIM * 3 * DIM + 255) / 256, 256, 0, stream>>>(
        W_O, W_I, W_S, b_O, b_I, b_S, Wcat, bias);

    // 3) edge scatter (both directions)
    long long ethreads = (long long)NE * 32;
    cgcn_edge_scatter<<<(unsigned)((ethreads + 255) / 256), 256, 0, stream>>>(
        node_embs, edge_embs, src, dst, sum_out, sum_in, cnt_out, cnt_in, (int)NE);

    // 4) fused mean-normalize + concat-GEMM (WMMA bf16) + bias + BN stats
    unsigned gblocks = (unsigned)((NN + 15) / 16);
    cgcn_gemm_wmma<<<gblocks, 256, 0, stream>>>(
        sum_out, sum_in, cnt_out, cnt_in, node_embs, Wcat, bias,
        out, stat_sum, stat_sq, (int)NN);

    // 5) BN finalize (scale/shift per feature)
    cgcn_bn_finalize<<<1, 128, 0, stream>>>(
        stat_sum, stat_sq, bn_gamma, bn_beta, scale, shift, 1.0f / (float)NN);

    // 6) BN apply in place on d_out
    long long n4 = (long long)NN * (DIM / 4);
    cgcn_bn_apply<<<(unsigned)((n4 + 255) / 256), 256, 0, stream>>>(out, scale, shift, n4);
}
